// ESupConLoss_40029095198731
// MI455X (gfx1250) — matrix operand order repaired and checked by reading, not compiled, not run on hardware
//
#include <hip/hip_runtime.h>

// ============================================================================
// ESupConLoss fused kernel for MI455X (gfx1250, wave32, WMMA).
//
// Roofline: 51.5 GFLOP of GEMM over an 8 MB input -> must fuse GEMM+exp+rowsum
// (never materialize the 3x 256MB similarity matrices). bf16 working set (4MB)
// fits in 192MB L2, so the fused kernel is WMMA/exp bound, not HBM bound.
// All exp's are < e^1, so sum-exp minus exact diagonal terms == masked
// logsumexp without needing a running max.
// ============================================================================

typedef __attribute__((ext_vector_type(16))) __bf16 v16bf;
typedef __attribute__((ext_vector_type(8)))  float  v8f;

#define DIM   128
#define ALPHA 2.0f

union Frag16 {
    v16bf v;
    uint4 q[2];
};

// ---- fp32 -> bf16 (round to nearest even) ----------------------------------
__device__ inline unsigned short f2bf(float f) {
    unsigned u = __float_as_uint(f);
    u += 0x7FFFu + ((u >> 16) & 1u);
    return (unsigned short)(u >> 16);
}

// ---- A-matrix fragment, 16x32 bf16 (ISA 7.12.2) ----------------------------
// lanes 0-15: M=lane, e0..7 -> K=k0..k0+7,  e8..15 -> K=k0+16..k0+23
// lanes16-31: M=lane-16, e0..7 -> K=k0+8..k0+15, e8..15 -> K=k0+24..k0+31
__device__ inline v16bf load_a_frag(const unsigned short* Z, int rowBase, int k0, int lane) {
    int m    = lane & 15;
    int koff = (lane >> 4) << 3;                 // 0 or 8
    const unsigned short* p = Z + (size_t)(rowBase + m) * DIM + k0 + koff;
    Frag16 f;
    f.q[0] = *(const uint4*)(p);                 // K = k0+koff .. +7
    f.q[1] = *(const uint4*)(p + 16);            // K = k0+16+koff .. +7
    return f.v;
}

// ---- B-matrix fragment, 32x16 bf16; B = Z^T so it's a contiguous row read --
// lanes 0-15: N=lane,    e0..7 -> K=k0..k0+7,   e8..15 -> K=k0+16..k0+23
// lanes16-31: N=lane-16, e0..7 -> K=k0+8..k0+15, e8..15 -> K=k0+24..k0+31
__device__ inline v16bf load_b_frag(const unsigned short* Z, int colBase, int k0, int lane) {
    int nrow = colBase + (lane & 15);
    int koff = (lane >> 4) << 3;                 // 0 or 8
    const unsigned short* p = Z + (size_t)nrow * DIM + k0 + koff;
    Frag16 f;
    f.q[0] = *(const uint4*)(p);
    f.q[1] = *(const uint4*)(p + 16);
    return f.v;
}

// ============================================================================
// Kernel 1: convert fp32 inputs to bf16 in workspace
// ============================================================================
__global__ __launch_bounds__(256) void convert_bf16_kernel(
    const float* __restrict__ za, const float* __restrict__ zt,
    unsigned short* __restrict__ Abf, unsigned short* __restrict__ Tbf, int total)
{
    int stride = gridDim.x * blockDim.x;
    for (int i = blockIdx.x * blockDim.x + threadIdx.x; i < total; i += stride) {
        Abf[i] = f2bf(za[i]);
        Tbf[i] = f2bf(zt[i]);
    }
}

// ============================================================================
// Kernel 2: exact fp32 diagonal terms + per-row pt contribution
// ============================================================================
__global__ __launch_bounds__(256) void diag_kernel(
    const float* __restrict__ za, const float* __restrict__ zt,
    const float* __restrict__ fc,
    float* __restrict__ exp_aa, float* __restrict__ exp_tt,
    float* __restrict__ exp_at, float* __restrict__ diag_at,
    float* __restrict__ ptrow, int n)
{
    int i = blockIdx.x * blockDim.x + threadIdx.x;
    if (i >= n) return;
    const float* a = za + (size_t)i * DIM;
    const float* t = zt + (size_t)i * DIM;
    float saa = 0.f, stt = 0.f, sat = 0.f, pt = 0.f;
#pragma unroll 8
    for (int k = 0; k < DIM; k++) {
        float av = a[k], tv = t[k];
        saa += av * av;
        stt += tv * tv;
        sat += av * tv;
        pt  += (av - tv) * (fc[DIM + k] - fc[k]);   // <z_au - z_tp, pt1 - pt0>
    }
    exp_aa[i]  = __expf(saa);
    exp_tt[i]  = __expf(stt);
    exp_at[i]  = __expf(sat);
    diag_at[i] = sat;
    ptrow[i]   = pt;
}

// ============================================================================
// Kernel 3: fused WMMA similarity + exp row-sum. One wave owns 16 rows, holds
// A fragments (z_au & z_tp, K=128) in VGPRs, sweeps all column tiles.
// ============================================================================
__global__ __launch_bounds__(256) void simrow_kernel(
    const unsigned short* __restrict__ Abf, const unsigned short* __restrict__ Tbf,
    const float* __restrict__ exp_aa, const float* __restrict__ exp_tt,
    const float* __restrict__ exp_at, const float* __restrict__ diag_at,
    float* __restrict__ rowvals, int n)
{
    const int lane    = threadIdx.x & 31;
    const int wave    = threadIdx.x >> 5;
    const int rowBase = blockIdx.x * 128 + wave * 16;

    // A fragments for our 16 rows, K = 128 = 4 x 32 (both matrices)
    v16bf a_au[4], a_tp[4];
#pragma unroll
    for (int kb = 0; kb < 4; kb++) {
        a_au[kb] = load_a_frag(Abf, rowBase, kb * 32, lane);
        a_tp[kb] = load_a_frag(Tbf, rowBase, kb * 32, lane);
    }

    // per-lane running sums; element e belongs to row m = e + 8*(lane>=16)
    float acc_pos[8], acc_neg[8];
#pragma unroll
    for (int e = 0; e < 8; e++) { acc_pos[e] = 0.f; acc_neg[e] = 0.f; }

    for (int j0 = 0; j0 < n; j0 += 16) {
        v16bf b_au[4], b_tp[4];
#pragma unroll
        for (int kb = 0; kb < 4; kb++) {
            b_au[kb] = load_b_frag(Abf, j0, kb * 32, lane);
            b_tp[kb] = load_b_frag(Tbf, j0, kb * 32, lane);
        }
        v8f c_aa = {}, c_tt = {}, c_at = {};
#pragma unroll
        for (int kb = 0; kb < 4; kb++) {   // 3 independent chains hide WMMA latency
            c_aa = __builtin_amdgcn_wmma_f32_16x16x32_bf16(
                false, a_au[kb], false, b_au[kb], (short)0, c_aa, false, false);
            c_tt = __builtin_amdgcn_wmma_f32_16x16x32_bf16(
                false, a_tp[kb], false, b_tp[kb], (short)0, c_tt, false, false);
            c_at = __builtin_amdgcn_wmma_f32_16x16x32_bf16(
                false, a_au[kb], false, b_tp[kb], (short)0, c_at, false, false);
        }
#pragma unroll
        for (int e = 0; e < 8; e++) {
            acc_pos[e] += __expf(c_aa[e]) + __expf(c_tt[e]);
            acc_neg[e] += __expf(c_at[e]);
        }
    }

    // reduce each row across its 16-lane half (xor masks < 16 stay in-half)
#pragma unroll
    for (int e = 0; e < 8; e++) {
        float p = acc_pos[e], g = acc_neg[e];
#pragma unroll
        for (int m = 8; m >= 1; m >>= 1) {
            p += __shfl_xor(p, m, 32);
            g += __shfl_xor(g, m, 32);
        }
        acc_pos[e] = p; acc_neg[e] = g;
    }

    // lanes 0 and 16 own rows 0..7 / 8..15 of this tile
    if ((lane & 15) == 0) {
        int mbase = (lane >> 4) * 8;
#pragma unroll
        for (int e = 0; e < 8; e++) {
            int i = rowBase + mbase + e;
            float pos_sum = acc_pos[e] - exp_aa[i] - exp_tt[i];  // remove diagonal
            float neg_sum = acc_neg[e] - exp_at[i];
            rowvals[i] = -__logf(pos_sum) + __logf(neg_sum) + ALPHA * diag_at[i];
        }
    }
}

// ============================================================================
// Kernel 4: deterministic fixed-order reduction -> final scalar
// ============================================================================
__global__ __launch_bounds__(256) void reduce_kernel(
    const float* __restrict__ rowvals, const float* __restrict__ ptrow,
    float* __restrict__ out, int n)
{
    __shared__ float s1[256], s2[256];
    int tid = threadIdx.x;
    float r = 0.f, p = 0.f;
    for (int i = tid; i < n; i += 256) { r += rowvals[i]; p += ptrow[i]; }
    s1[tid] = r; s2[tid] = p;
    __syncthreads();
    for (int off = 128; off > 0; off >>= 1) {
        if (tid < off) { s1[tid] += s1[tid + off]; s2[tid] += s2[tid + off]; }
        __syncthreads();
    }
    if (tid == 0) {
        float nf = (float)n;
        float total = s1[0] + s2[0] / nf;            // supcon + pt_loss
        out[0] = (nf * (float)DIM) / (nf + 2.0f) * total;
    }
}

// ============================================================================
extern "C" void kernel_launch(void* const* d_in, const int* in_sizes, int n_in,
                              void* d_out, int out_size, void* d_ws, size_t ws_size,
                              hipStream_t stream) {
    const float* z_au = (const float*)d_in[0];
    const float* z_tp = (const float*)d_in[1];
    const float* fc   = (const float*)d_in[2];
    float* out = (float*)d_out;

    const int n = in_sizes[0] / DIM;        // 8192
    const int total = n * DIM;

    // workspace layout (~4.2 MB)
    unsigned short* Abf = (unsigned short*)d_ws;          // n*DIM bf16
    unsigned short* Tbf = Abf + (size_t)total;            // n*DIM bf16
    float* exp_aa  = (float*)(Tbf + (size_t)total);
    float* exp_tt  = exp_aa + n;
    float* exp_at  = exp_tt + n;
    float* diag_at = exp_at + n;
    float* ptrow   = diag_at + n;
    float* rowvals = ptrow + n;

    convert_bf16_kernel<<<512, 256, 0, stream>>>(z_au, z_tp, Abf, Tbf, total);
    diag_kernel<<<(n + 255) / 256, 256, 0, stream>>>(
        z_au, z_tp, fc, exp_aa, exp_tt, exp_at, diag_at, ptrow, n);
    simrow_kernel<<<n / 128, 256, 0, stream>>>(
        Abf, Tbf, exp_aa, exp_tt, exp_at, diag_at, rowvals, n);
    reduce_kernel<<<1, 256, 0, stream>>>(rowvals, ptrow, out, n);
}
